// RNN_model_68513318306089
// MI455X (gfx1250) — compile-verified, matrix operand order checked by reading, not run on hardware
//
#include <hip/hip_runtime.h>
#include <hip/hip_bf16.h>
#include <cstdint>
#include <cstddef>

// ---------------------------------------------------------------- constants
#define VOCAB 32000
#define EMB   512
#define HID   1024
#define BATCH 16
#define TSEQ  128
#define NG    (4*HID)   // 4096 fused gate columns (i,f,o,c)

#define LDS_STRIDE 1032 // 16-byte aligned row stride; 516 words % 64 banks == 4 -> conflict-free b128

typedef __attribute__((ext_vector_type(16))) __bf16 v16bf;
typedef __attribute__((ext_vector_type(8)))  float  v8f;

// float -> bf16 with round-to-nearest-even
static __device__ __forceinline__ uint16_t f2bf(float f) {
    uint32_t u = __float_as_uint(f);
    uint32_t r = (u + 0x7FFFu + ((u >> 16) & 1u)) >> 16;
    return (uint16_t)r;
}

// Load one 16x32 bf16 WMMA fragment slice for this lane (global or LDS pointer).
// koff = 0 for lanes 0..15, 8 for lanes 16..31 (per CDNA5 16-bit A/B layout).
union Frag32 { v16bf v; int4 q[2]; };
static __device__ __forceinline__ v16bf load_frag(const uint16_t* base, int koff) {
    Frag32 f;
    f.q[0] = *(const int4*)(base + koff);        // K = koff .. koff+7
    f.q[1] = *(const int4*)(base + koff + 16);   // K = koff+16 .. koff+23
    return f.v;
}

static __device__ __forceinline__ v8f wmma_bf16(v16bf a, v16bf b, v8f c) {
    return __builtin_amdgcn_wmma_f32_16x16x32_bf16(false, a, false, b, (short)0, c,
                                                   false, false);
}

// ---------------------------------------------------------------- kernels

// Xbf[m=t*B+b][e] = bf16(emb[sentence[b][t]][e])
__global__ void k_embed(const int* __restrict__ sent, const float* __restrict__ emb,
                        uint16_t* __restrict__ Xbf) {
    int id = blockIdx.x * blockDim.x + threadIdx.x;
    if (id >= BATCH * TSEQ * EMB) return;
    int e = id & (EMB - 1);
    int m = id >> 9;               // /EMB
    int t = m >> 4, b = m & 15;
    int tok = sent[b * TSEQ + t];
    Xbf[id] = f2bf(emb[(size_t)tok * EMB + e]);
}

// Transpose + convert the 4 gate weight matrices into [n=g*H+j][k] bf16.
// WxT: k<EMB rows of W (input part), WhT: k>=EMB rows (recurrent part).
__global__ void k_wgate(const float* __restrict__ Wi, const float* __restrict__ Wf,
                        const float* __restrict__ Wo, const float* __restrict__ Wc,
                        uint16_t* __restrict__ WxT, uint16_t* __restrict__ WhT) {
    int id = blockIdx.x * blockDim.x + threadIdx.x;
    if (id >= NG * (EMB + HID)) return;
    int k = id % (EMB + HID);
    int n = id / (EMB + HID);
    int g = n >> 10, j = n & (HID - 1);
    const float* Wg = (g == 0) ? Wi : (g == 1) ? Wf : (g == 2) ? Wo : Wc;
    uint16_t v = f2bf(Wg[(size_t)k * HID + j]);
    if (k < EMB) WxT[(size_t)n * EMB + k] = v;
    else         WhT[(size_t)n * HID + (k - EMB)] = v;
}

// WhqT[v][k] = bf16(W_hq[k][v])
__global__ void k_whq(const float* __restrict__ Whq, uint16_t* __restrict__ WhqT) {
    int id = blockIdx.x * blockDim.x + threadIdx.x;
    if (id >= HID * VOCAB) return;
    int v = id % VOCAB;
    int k = id / VOCAB;
    WhqT[(size_t)v * HID + k] = f2bf(Whq[(size_t)k * VOCAB + v]);
}

// zero h0 (bf16) and c state (fp32)
__global__ void k_init(uint16_t* __restrict__ h0, float* __restrict__ c) {
    int id = blockIdx.x * blockDim.x + threadIdx.x;
    if (id >= BATCH * HID) return;
    h0[id] = 0;
    c[id]  = 0.0f;
}

// Zx = X @ Wx   M=2048, N=4096, K=512.
// One wave computes 1 M-tile x 4 N-tiles (A fragment reused 4x -> 4x less A L2 traffic).
__global__ void k_zx(const uint16_t* __restrict__ Xbf, const uint16_t* __restrict__ WxT,
                     float* __restrict__ Zx) {
    int wid  = (blockIdx.x * blockDim.x + threadIdx.x) >> 5;   // 0..8191
    int lane = threadIdx.x & 31;
    int mtile  = wid >> 6;    // 0..127
    int ngroup = wid & 63;    // 0..63  (64 columns each)
    int koff = (lane < 16) ? 0 : 8;
    const uint16_t* arow = Xbf + (size_t)(mtile * 16 + (lane & 15)) * EMB;
    const uint16_t* bcol = WxT + (size_t)(ngroup * 64 + (lane & 15)) * EMB;
    v8f acc[4] = {{}, {}, {}, {}};
    #pragma unroll 2
    for (int kk = 0; kk < EMB; kk += 32) {
        v16bf a = load_frag(arow + kk, koff);
        #pragma unroll
        for (int q = 0; q < 4; ++q) {
            v16bf b = load_frag(bcol + (size_t)q * 16 * EMB + kk, koff);
            acc[q] = wmma_bf16(a, b, acc[q]);
        }
    }
    int mbase = mtile * 16 + ((lane < 16) ? 0 : 8);
    #pragma unroll
    for (int q = 0; q < 4; ++q) {
        int n = ngroup * 64 + q * 16 + (lane & 15);
        #pragma unroll
        for (int r = 0; r < 8; ++r)
            Zx[(size_t)(mbase + r) * NG + n] = acc[q][r];
    }
}

// One LSTM time step. grid = 64 blocks (16-wide hidden tiles), block = 32 (1 wave).
// hprev: [B][HID] bf16 (row t-1 of hs, or h0), zrow: Zx row block for this t,
// writes c state and h_t into hs (also consumed as next step's A).
__global__ void k_step(const uint16_t* __restrict__ WhT, const float* __restrict__ zrow,
                       const float* __restrict__ bi, const float* __restrict__ bf_,
                       const float* __restrict__ bo, const float* __restrict__ bc,
                       const uint16_t* __restrict__ hprev, float* __restrict__ cst,
                       uint16_t* __restrict__ hout) {
    int lane = threadIdx.x & 31;
    int jt   = blockIdx.x;                       // hidden tile 0..63
    int koff = (lane < 16) ? 0 : 8;
    const uint16_t* arow = hprev + (size_t)(lane & 15) * HID;
    const uint16_t* b0 = WhT + (size_t)(0 * HID + jt * 16 + (lane & 15)) * HID;
    const uint16_t* b1 = WhT + (size_t)(1 * HID + jt * 16 + (lane & 15)) * HID;
    const uint16_t* b2 = WhT + (size_t)(2 * HID + jt * 16 + (lane & 15)) * HID;
    const uint16_t* b3 = WhT + (size_t)(3 * HID + jt * 16 + (lane & 15)) * HID;
    v8f a0 = {}, a1 = {}, a2 = {}, a3 = {};
    #pragma unroll 2
    for (int kk = 0; kk < HID; kk += 32) {
        v16bf a = load_frag(arow + kk, koff);    // shared A across the 4 gates
        a0 = wmma_bf16(a, load_frag(b0 + kk, koff), a0);
        a1 = wmma_bf16(a, load_frag(b1 + kk, koff), a1);
        a2 = wmma_bf16(a, load_frag(b2 + kk, koff), a2);
        a3 = wmma_bf16(a, load_frag(b3 + kk, koff), a3);
    }
    int j = jt * 16 + (lane & 15);
    int mbase = (lane < 16) ? 0 : 8;
    float vbi = bi[j], vbf = bf_[j], vbo = bo[j], vbc = bc[j];
    #pragma unroll
    for (int r = 0; r < 8; ++r) {
        int b = mbase + r;                       // batch index
        const float* zr = zrow + (size_t)b * NG;
        float zi = a0[r] + zr[0 * HID + j] + vbi;
        float zf = a1[r] + zr[1 * HID + j] + vbf;
        float zo = a2[r] + zr[2 * HID + j] + vbo;
        float zc = a3[r] + zr[3 * HID + j] + vbc;
        float ig = 1.0f / (1.0f + __expf(-zi));
        float fg = 1.0f / (1.0f + __expf(-zf));
        float og = 1.0f / (1.0f + __expf(-zo));
        float gg = tanhf(zc);
        float cn = fg * cst[b * HID + j] + ig * gg;
        cst[b * HID + j] = cn;
        hout[(size_t)b * HID + j] = f2bf(og * tanhf(cn));
    }
}

// Projection: out[b][v][t] = hs[t*B+b] . Whq[:,v] + bq[v]
// M=2048, N=32000, K=1024.
// Block = 4 waves: stages 2 M-tiles of A (32 rows x 1024 K, 66KB padded) into LDS once,
// each wave computes 2 M-tiles x 4 N-tiles (B fragments reused across both M-tiles).
__global__ void __launch_bounds__(128) k_proj(const uint16_t* __restrict__ hs,
                                              const uint16_t* __restrict__ WhqT,
                                              const float* __restrict__ bq,
                                              float* __restrict__ out) {
    extern __shared__ uint16_t As[];             // [32][LDS_STRIDE] bf16
    int tid   = threadIdx.x;
    int mpair = blockIdx.x / 125;                // 0..63  -> rows m0 .. m0+31
    int ngblk = blockIdx.x % 125;                // 0..124 -> 4 n-groups of 64 cols
    int m0 = mpair * 32;

    // stage A: 32 rows x 1024 bf16 -> LDS (coalesced b128 global loads, b128 ds stores)
    for (int i = tid; i < 32 * 128; i += 128) {  // 128 chunks of 8 elements per row
        int row = i >> 7;
        int c8  = (i & 127) << 3;
        *(int4*)(&As[row * LDS_STRIDE + c8]) =
            *(const int4*)(&hs[(size_t)(m0 + row) * HID + c8]);
    }
    __syncthreads();

    int wave = tid >> 5;
    int lane = tid & 31;
    int ngroup = ngblk * 4 + wave;               // 0..499 (64 columns each)
    int koff = (lane < 16) ? 0 : 8;
    const uint16_t* a0p = &As[(lane & 15) * LDS_STRIDE];          // M-tile 0
    const uint16_t* a1p = &As[((lane & 15) + 16) * LDS_STRIDE];   // M-tile 1
    const uint16_t* bcol = WhqT + (size_t)(ngroup * 64 + (lane & 15)) * HID;

    v8f acc[2][4] = {{{}, {}, {}, {}}, {{}, {}, {}, {}}};
    #pragma unroll 2
    for (int kk = 0; kk < HID; kk += 32) {
        v16bf a0 = load_frag(a0p + kk, koff);
        v16bf a1 = load_frag(a1p + kk, koff);
        #pragma unroll
        for (int q = 0; q < 4; ++q) {
            const uint16_t* bp = bcol + (size_t)q * 16 * HID;
            __builtin_prefetch(bp + kk + 256, 0, 3);   // global_prefetch_b8, 8 iters ahead
            v16bf b = load_frag(bp + kk, koff);
            acc[0][q] = wmma_bf16(a0, b, acc[0][q]);   // B fragment reused across M-tiles
            acc[1][q] = wmma_bf16(a1, b, acc[1][q]);
        }
    }

    #pragma unroll
    for (int q = 0; q < 4; ++q) {
        int v = ngroup * 64 + q * 16 + (lane & 15);
        float bias = bq[v];
        #pragma unroll
        for (int mt = 0; mt < 2; ++mt) {
            int mbase = m0 + mt * 16 + ((lane < 16) ? 0 : 8);
            #pragma unroll
            for (int r = 0; r < 8; ++r) {
                int m = mbase + r;
                int tt = m >> 4, bb = m & 15;    // m = t*B + b
                out[((size_t)bb * VOCAB + v) * TSEQ + tt] = acc[mt][q][r] + bias;
            }
        }
    }
}

// ---------------------------------------------------------------- launcher
extern "C" void kernel_launch(void* const* d_in, const int* in_sizes, int n_in,
                              void* d_out, int out_size, void* d_ws, size_t ws_size,
                              hipStream_t stream) {
    (void)in_sizes; (void)n_in; (void)out_size; (void)ws_size;
    const int*   sent = (const int*)  d_in[0];
    const float* emb  = (const float*)d_in[1];
    const float* Wi   = (const float*)d_in[2];
    const float* bi   = (const float*)d_in[3];
    const float* Wf   = (const float*)d_in[4];
    const float* bf_  = (const float*)d_in[5];
    const float* Wo   = (const float*)d_in[6];
    const float* bo   = (const float*)d_in[7];
    const float* Wc   = (const float*)d_in[8];
    const float* bc   = (const float*)d_in[9];
    const float* Whq  = (const float*)d_in[10];
    const float* bq   = (const float*)d_in[11];
    float* out = (float*)d_out;

    // carve workspace (all 256B aligned)
    char* p = (char*)d_ws;
    auto carve = [&](size_t bytes) -> void* {
        void* r = (void*)p;
        p += (bytes + 255) & ~(size_t)255;
        return r;
    };
    uint16_t* Xbf  = (uint16_t*)carve((size_t)TSEQ * BATCH * EMB * 2);   //   2 MB
    uint16_t* WxT  = (uint16_t*)carve((size_t)NG * EMB * 2);             //   4 MB
    uint16_t* WhT  = (uint16_t*)carve((size_t)NG * HID * 2);             //   8 MB
    uint16_t* WhqT = (uint16_t*)carve((size_t)VOCAB * HID * 2);          //  64 MB
    float*    Zx   = (float*)   carve((size_t)TSEQ * BATCH * NG * 4);    //  32 MB
    uint16_t* hs   = (uint16_t*)carve((size_t)TSEQ * BATCH * HID * 2);   //   4 MB
    uint16_t* h0   = (uint16_t*)carve((size_t)BATCH * HID * 2);
    float*    cst  = (float*)   carve((size_t)BATCH * HID * 4);

    // 1) embedding gather + bf16 convert
    k_embed<<<(BATCH * TSEQ * EMB + 255) / 256, 256, 0, stream>>>(sent, emb, Xbf);
    // 2) gate weight transpose (split input/recurrent parts)
    k_wgate<<<(NG * (EMB + HID) + 255) / 256, 256, 0, stream>>>(Wi, Wf, Wo, Wc, WxT, WhT);
    // 3) projection weight transpose
    k_whq<<<(HID * VOCAB + 255) / 256, 256, 0, stream>>>(Whq, WhqT);
    // 4) zero initial h and c
    k_init<<<(BATCH * HID + 255) / 256, 256, 0, stream>>>(h0, cst);
    // 5) hoisted input-side matmul: Zx = X @ Wx (2048x512x4096), 1 wave = 1Mx4N tiles
    k_zx<<<(128 * 64) / 8, 256, 0, stream>>>(Xbf, WxT, Zx);
    // 6) 128 sequential LSTM steps (stream order = recurrence sync)
    for (int t = 0; t < TSEQ; ++t) {
        const uint16_t* hprev = (t == 0) ? h0 : hs + (size_t)(t - 1) * BATCH * HID;
        k_step<<<HID / 16, 32, 0, stream>>>(
            WhT, Zx + (size_t)t * BATCH * NG, bi, bf_, bo, bc,
            hprev, cst, hs + (size_t)t * BATCH * HID);
    }
    // 7) output projection (2048x1024x32000): 64 Mpairs x 125 blocks, 4 waves/block,
    //    dynamic LDS = 32 rows * LDS_STRIDE * 2B = 66048 bytes (within 320KB WGP LDS)
    k_proj<<<64 * 125, 128, 32 * LDS_STRIDE * 2, stream>>>(hs, WhqT, bq, out);
}
